// AffinityNet_19894288515079
// MI455X (gfx1250) — compile-verified, matrix operand order checked by reading, not run on hardware
//
#include <hip/hip_runtime.h>
#include <hip/hip_bf16.h>
#include <math.h>

#define N_NODES 50000
#define N_PAD   50048          // 391 * 128 : padded row count for GEMM tiles
#define N_EDGES 600000
#define NF      128
#define N_GRAPHS 64

typedef __attribute__((ext_vector_type(16))) __bf16 v16bf;
typedef __attribute__((ext_vector_type(8)))  float  v8f;

// ---------------------------------------------------------------------------
// Degree / normalization:  deg[n] = 1 (self loop) + sum_{e: dst==n} ew[e]
// then dinv[n] = rsqrt(deg)
// ---------------------------------------------------------------------------
__global__ void k_deg_init(float* deg) {
    int n = blockIdx.x * blockDim.x + threadIdx.x;
    if (n < N_NODES) deg[n] = 1.0f;
}

__global__ void k_deg_edges(const int* __restrict__ dst,
                            const float* __restrict__ ew,
                            float* __restrict__ deg) {
    int e = blockIdx.x * blockDim.x + threadIdx.x;
    if (e < N_EDGES) atomicAdd(&deg[dst[e]], ew[e]);
}

__global__ void k_deg_fin(float* deg) {
    int n = blockIdx.x * blockDim.x + threadIdx.x;
    if (n < N_NODES) {
        float d = deg[n];
        deg[n] = (d > 0.0f) ? rsqrtf(d) : 0.0f;   // deg buffer becomes dinv
    }
}

// ---------------------------------------------------------------------------
// GEMM: C[N_PAD x 128] = A[Mload x 128] @ W[128 x 128]
// fp32 -> bf16 -> v_wmma_f32_16x16x32_bf16, f32 accumulate.
// Block = 256 threads = 8 waves, tile 128 rows x 128 cols.
// W staged in LDS pre-swizzled into B-fragment order: fragment
// (nt,ks) = 32 lanes x 16 contiguous bf16 -> one 32B LDS vector load per lane.
// OOB rows are redirected to a zeroed pad row (no per-element selects);
// stores are unconditional (C always has N_PAD = gridDim.x*128 rows).
// ---------------------------------------------------------------------------
__global__ __launch_bounds__(256) void k_gemm128(const float* __restrict__ A,
                                                 const float* __restrict__ W,
                                                 float* __restrict__ C, int Mload,
                                                 const float* __restrict__ zrow) {
    __shared__ v16bf Wswz[1024];           // 8 nt * 4 ks * 32 lanes, 32 KB
    __bf16* wp = (__bf16*)Wswz;

    const int t = threadIdx.x;
    // Pre-swizzle W into B-fragment layout.
    // Element e of fragment (nt,ks,lane):  n = nt*16 + (lane&15), hi = lane>>4,
    //   k = ks*32 + (e<8 ? e : e+8) + hi*8
    for (int idx = t; idx < 16384; idx += 256) {
        int e    = idx & 15;
        int fid  = idx >> 4;
        int lane = fid & 31;
        int ks   = (fid >> 5) & 3;
        int nt   = fid >> 7;
        int n  = nt * 16 + (lane & 15);
        int hi = lane >> 4;
        int k  = ks * 32 + (e < 8 ? e : e + 8) + hi * 8;
        wp[idx] = (__bf16)W[k * 128 + n];
    }
    __syncthreads();

    const int wave = t >> 5;               // 0..7
    const int lane = t & 31;
    const int mr   = lane & 15;
    const int hi   = lane >> 4;
    const int m0   = blockIdx.x * 128 + wave * 16;   // wave's 16-row strip
    const int row  = m0 + mr;
    // OOB lanes read a zeroed 128-float pad row: no per-element selects.
    const float4* ar4 = reinterpret_cast<const float4*>(
        (row < Mload) ? (A + (size_t)row * NF) : zrow);

    // 4 A fragments covering K = 0..127 ; per k-step the lane's 16 elements
    // are two contiguous 8-float runs: [k0, k0+8) and [k0+16, k0+24),
    // k0 = ks*32 + hi*8  ->  4x float4 vector loads.
    v16bf afrag[4];
#pragma unroll
    for (int ks = 0; ks < 4; ++ks) {
        int q = (ks * 32 + hi * 8) >> 2;   // float4 index of first run
        float4 a0 = ar4[q];
        float4 a1 = ar4[q + 1];
        float4 a2 = ar4[q + 4];
        float4 a3 = ar4[q + 5];
        float f[16] = {a0.x, a0.y, a0.z, a0.w, a1.x, a1.y, a1.z, a1.w,
                       a2.x, a2.y, a2.z, a2.w, a3.x, a3.y, a3.z, a3.w};
#pragma unroll
        for (int e = 0; e < 16; ++e)
            afrag[ks][e] = (__bf16)f[e];
    }

    // 8 output n-tiles of 16 cols each
#pragma unroll
    for (int nt = 0; nt < 8; ++nt) {
        v8f acc = {0.f, 0.f, 0.f, 0.f, 0.f, 0.f, 0.f, 0.f};
#pragma unroll
        for (int ks = 0; ks < 4; ++ks) {
            v16bf bfrag = Wswz[(nt * 4 + ks) * 32 + lane];
            acc = __builtin_amdgcn_wmma_f32_16x16x32_bf16(
                /*neg_a=*/false, afrag[ks], /*neg_b=*/false, bfrag,
                /*c_mod=*/(short)0, acc, /*reuse_a=*/false, /*reuse_b=*/false);
        }
        // C/D layout: VGPR r -> row m0 + hi*8 + r, col nt*16 + (lane&15)
        float* crow = C + (size_t)(m0 + hi * 8) * NF + nt * 16 + mr;
#pragma unroll
        for (int r = 0; r < 8; ++r)
            crow[(size_t)r * NF] = acc[r];
    }
}

// ---------------------------------------------------------------------------
// Self-loop init: Aacc[n] = H[n] * dinv[n]^2   (one wave per node, float4/lane)
// ---------------------------------------------------------------------------
__global__ void k_selfloop(const float* __restrict__ H,
                           const float* __restrict__ dinv,
                           float* __restrict__ Aacc) {
    int tid  = blockIdx.x * blockDim.x + threadIdx.x;
    int n    = tid >> 5;
    int lane = tid & 31;
    if (n >= N_NODES) return;
    float di = dinv[n];
    float s  = di * di;
    float4 h = reinterpret_cast<const float4*>(H + (size_t)n * NF)[lane];
    float4 o = {h.x * s, h.y * s, h.z * s, h.w * s};
    reinterpret_cast<float4*>(Aacc + (size_t)n * NF)[lane] = o;
}

// ---------------------------------------------------------------------------
// Edge scatter: Aacc[dst] += H[src] * (dinv[src]*ew*dinv[dst])
// one wave per edge iteration, float4 gather + 4x atomicAdd per lane.
// Prefetch next iteration's source row (global_prefetch_b8 path).
// ---------------------------------------------------------------------------
__global__ void k_edge_scatter(const int* __restrict__ src,
                               const int* __restrict__ dst,
                               const float* __restrict__ ew,
                               const float* __restrict__ dinv,
                               const float* __restrict__ H,
                               float* __restrict__ Aacc) {
    int tid    = blockIdx.x * blockDim.x + threadIdx.x;
    int wave   = tid >> 5;
    int lane   = tid & 31;
    int nwaves = (gridDim.x * blockDim.x) >> 5;
    for (int e = wave; e < N_EDGES; e += nwaves) {
        int s = src[e];
        int d = dst[e];
        // prefetch the next gathered row while this one is in flight
        int e2 = e + nwaves;
        if (e2 < N_EDGES) {
            int s2 = src[e2];
            __builtin_prefetch(H + (size_t)s2 * NF + lane * 4, 0, 0);
        }
        float norm = dinv[s] * ew[e] * dinv[d];
        float4 h = reinterpret_cast<const float4*>(H + (size_t)s * NF)[lane];
        float* outp = Aacc + (size_t)d * NF + lane * 4;
        atomicAdd(outp + 0, h.x * norm);
        atomicAdd(outp + 1, h.y * norm);
        atomicAdd(outp + 2, h.z * norm);
        atomicAdd(outp + 3, h.w * norm);
    }
}

// ---------------------------------------------------------------------------
// Bias + ReLU (in place), float4 per lane
// ---------------------------------------------------------------------------
__global__ void k_bias_relu(float* __restrict__ Aacc, const float* __restrict__ b) {
    int tid = blockIdx.x * blockDim.x + threadIdx.x;
    if (tid >= N_NODES * 32) return;
    int lane = tid & 31;
    float4 v  = reinterpret_cast<float4*>(Aacc)[tid];
    float4 bb = reinterpret_cast<const float4*>(b)[lane];
    v.x = fmaxf(v.x + bb.x, 0.0f);
    v.y = fmaxf(v.y + bb.y, 0.0f);
    v.z = fmaxf(v.z + bb.z, 0.0f);
    v.w = fmaxf(v.w + bb.w, 0.0f);
    reinterpret_cast<float4*>(Aacc)[tid] = v;
}

// ---------------------------------------------------------------------------
// Zero helper
// ---------------------------------------------------------------------------
__global__ void k_zero(float* p, int n) {
    int i = blockIdx.x * blockDim.x + threadIdx.x;
    if (i < n) p[i] = 0.0f;
}

// ---------------------------------------------------------------------------
// Global mean pool (accumulate): gsum[batch[n]] += act[n]; gcnt[batch[n]] += 1
// ---------------------------------------------------------------------------
__global__ void k_pool(const float* __restrict__ act,
                       const int* __restrict__ batch,
                       float* __restrict__ gsum,
                       float* __restrict__ gcnt) {
    int tid    = blockIdx.x * blockDim.x + threadIdx.x;
    int wave   = tid >> 5;
    int lane   = tid & 31;
    int nwaves = (gridDim.x * blockDim.x) >> 5;
    for (int n = wave; n < N_NODES; n += nwaves) {
        int g = batch[n];
        float4 a = reinterpret_cast<const float4*>(act + (size_t)n * NF)[lane];
        float* gp = gsum + (size_t)g * NF + lane * 4;
        atomicAdd(gp + 0, a.x);
        atomicAdd(gp + 1, a.y);
        atomicAdd(gp + 2, a.z);
        atomicAdd(gp + 3, a.w);
        if (lane == 0) atomicAdd(&gcnt[g], 1.0f);
    }
}

// ---------------------------------------------------------------------------
// Head: g = gsum/max(cnt,1); z = relu(g@fc1W + fc1b); BatchNorm (batch stats,
// biased var); out = z@fc3W + fc3b.  Single block, all in LDS.
// ---------------------------------------------------------------------------
__global__ __launch_bounds__(256) void k_head(const float* __restrict__ gsum,
                                              const float* __restrict__ gcnt,
                                              const float* __restrict__ fc1W,  // 128x64
                                              const float* __restrict__ fc1b,  // 64
                                              const float* __restrict__ gamma,
                                              const float* __restrict__ beta,
                                              const float* __restrict__ fc3W,  // 64
                                              const float* __restrict__ fc3b,  // 1
                                              float* __restrict__ out) {       // 64
    __shared__ float g[N_GRAPHS * NF];     // 32 KB
    __shared__ float z[N_GRAPHS * 64];     // 16 KB
    __shared__ float mean[64], var[64];
    int t = threadIdx.x;

    for (int i = t; i < N_GRAPHS * NF; i += 256) {
        float c = gcnt[i >> 7];
        c = c > 1.0f ? c : 1.0f;
        g[i] = gsum[i] / c;
    }
    __syncthreads();

    for (int i = t; i < N_GRAPHS * 64; i += 256) {
        int r = i >> 6, c = i & 63;
        float s = fc1b[c];
        for (int k = 0; k < NF; ++k) s += g[r * NF + k] * fc1W[k * 64 + c];
        z[i] = s > 0.0f ? s : 0.0f;
    }
    __syncthreads();

    if (t < 64) {
        float m = 0.0f;
        for (int r = 0; r < N_GRAPHS; ++r) m += z[r * 64 + t];
        m *= (1.0f / N_GRAPHS);
        float v = 0.0f;
        for (int r = 0; r < N_GRAPHS; ++r) {
            float d = z[r * 64 + t] - m;
            v += d * d;
        }
        v *= (1.0f / N_GRAPHS);
        mean[t] = m;
        var[t]  = v;
    }
    __syncthreads();

    if (t < N_GRAPHS) {
        float s = 0.0f;
        for (int c = 0; c < 64; ++c) {
            float zn = gamma[c] * (z[t * 64 + c] - mean[c]) * rsqrtf(var[c] + 1e-5f) + beta[c];
            s += zn * fc3W[c];
        }
        out[t] = s + fc3b[0];
    }
}

// ---------------------------------------------------------------------------
// Host launcher
// ---------------------------------------------------------------------------
extern "C" void kernel_launch(void* const* d_in, const int* in_sizes, int n_in,
                              void* d_out, int out_size, void* d_ws, size_t ws_size,
                              hipStream_t stream) {
    const float* x     = (const float*)d_in[0];
    const int*   ei    = (const int*)d_in[1];    // [2, E] flattened
    const int*   batch = (const int*)d_in[2];
    const float* ew    = (const float*)d_in[3];
    const float* W1    = (const float*)d_in[4];
    const float* b1    = (const float*)d_in[5];
    const float* W2    = (const float*)d_in[6];
    const float* b2    = (const float*)d_in[7];
    const float* fc1W  = (const float*)d_in[8];
    const float* fc1b  = (const float*)d_in[9];
    const float* gam   = (const float*)d_in[10];
    const float* bet   = (const float*)d_in[11];
    const float* fc3W  = (const float*)d_in[12];
    const float* fc3b  = (const float*)d_in[13];
    float* out = (float*)d_out;

    const int* src = ei;
    const int* dst = ei + N_EDGES;

    // workspace layout (floats) ; H and Aacc padded to N_PAD rows so the
    // GEMM can store full 128-row tiles unconditionally.
    float* ws   = (float*)d_ws;
    float* dinv = ws;                                 // 50048 (deg -> dinv)
    float* H    = dinv + N_PAD;                       // N_PAD*128
    float* Aacc = H + (size_t)N_PAD * NF;             // N_PAD*128
    float* gsum = Aacc + (size_t)N_PAD * NF;          // 64*128
    float* gcnt = gsum + N_GRAPHS * NF;               // 64
    float* zrow = gcnt + N_GRAPHS;                    // 128 (zero pad row)

    const int T = 256;
    const int gemm_blocks = N_PAD / 128;              // 391

    // zero pad row (used by GEMM OOB lanes)
    k_zero<<<1, NF, 0, stream>>>(zrow, NF);

    // degree / dinv (shared by both conv layers)
    k_deg_init <<<(N_NODES + T - 1) / T, T, 0, stream>>>(dinv);
    k_deg_edges<<<(N_EDGES + T - 1) / T, T, 0, stream>>>(dst, ew, dinv);
    k_deg_fin  <<<(N_NODES + T - 1) / T, T, 0, stream>>>(dinv);

    // layer 1: H = x@W1 ; Aacc = aggregate(H) ; relu(Aacc + b1)
    k_gemm128<<<gemm_blocks, T, 0, stream>>>(x, W1, H, N_NODES, zrow);
    k_selfloop<<<(N_NODES * 32 + T - 1) / T, T, 0, stream>>>(H, dinv, Aacc);
    k_edge_scatter<<<4096, T, 0, stream>>>(src, dst, ew, dinv, H, Aacc);
    k_bias_relu<<<(N_NODES * 32 + T - 1) / T, T, 0, stream>>>(Aacc, b1);

    // layer 2: H = Aacc@W2 ; Aacc = aggregate(H) ; relu(Aacc + b2)
    // Aacc is padded, so Mload = N_PAD makes all loads unconditional too.
    k_gemm128<<<gemm_blocks, T, 0, stream>>>(Aacc, W2, H, N_PAD, zrow);
    k_selfloop<<<(N_NODES * 32 + T - 1) / T, T, 0, stream>>>(H, dinv, Aacc);
    k_edge_scatter<<<4096, T, 0, stream>>>(src, dst, ew, dinv, H, Aacc);
    k_bias_relu<<<(N_NODES * 32 + T - 1) / T, T, 0, stream>>>(Aacc, b2);

    // pool + head
    k_zero<<<(N_GRAPHS * NF + N_GRAPHS + T - 1) / T, T, 0, stream>>>(gsum, N_GRAPHS * NF + N_GRAPHS);
    k_pool<<<2048, T, 0, stream>>>(Aacc, batch, gsum, gcnt);
    k_head<<<1, T, 0, stream>>>(gsum, gcnt, fc1W, fc1b, gam, bet, fc3W, fc3b, out);
}
